// SpaceTransformer_elsa_63814624084302
// MI455X (gfx1250) — compile-verified, matrix operand order checked by reading, not run on hardware
//
#include <hip/hip_runtime.h>
#include <hip/hip_bf16.h>
#include <math.h>

typedef __attribute__((ext_vector_type(16))) _Float16 v16h;
typedef __attribute__((ext_vector_type(8)))  float    v8f;
typedef __attribute__((ext_vector_type(2)))  _Float16 h2;

#define CCH   96      // channels
#define NH    6
#define HDIM  16      // head dim == WMMA N tile
#define KS    7
#define K2    49
#define TT    16      // spatial tile
#define HALO  3
#define SS    22      // TT + 2*HALO
#define SP    484     // SS*SS
#define SROWS 512     // padded halo rows (31 m-tiles cover 496)
#define FRW   96
#define SCL   0.25f   // HDIM^-0.5
#define EPSL  1e-5f
#define CH_STRIDE 73728  // 8*96*96 (channel stride in x)

// packed weight layout in d_ws (halves): GEMM frags = 512 halves = [lane(32)][e(16)]
#define NT_QKV 18
#define WS_QKV 0
#define QKV_HALVES (NT_QKV*3*512)                 // 27648
#define WS_A2  QKV_HALVES
#define A2_HALVES (NH*4*3*512)                    // 36864 (per-head 4 ntiles padded)
#define WS_P   (WS_A2 + A2_HALVES)                // 64512
#define P_HALVES (6*3*512)                        // 9216
#define WS_A1  (WS_P + P_HALVES)                  // 73728: conv weights [oc][tap][ic] f16
#define A1_HALVES (CCH*K2*4)                      // 18816
#define WS_TOTAL (WS_A1 + A1_HALVES)              // 92544 halves = 185088 B

// LDS regions in halves (total 147456 halves = 294912 B dynamic LDS)
#define LDS_XN   0                 // xn[512][96]; later a[256][96] + attn[256][64]
#define LDS_A    0
#define LDS_ATTN 24576
#define LDS_H    49152             // h[512][96]; later agg[256][96]
#define LDS_AGG  49152
#define LDS_V    98304             // v[512][96]
#define SMEM_BYTES (147456*2)

// K index inside a 16-bit A/B fragment (ISA 7.12.2): lanes 0-15 get K 0-7,16-23;
// lanes 16-31 get K 8-15,24-31; element e packs 2 halves per VGPR.
__device__ __forceinline__ int kfrag(int lane, int e) {
  return ((e < 8) ? e : (e + 8)) + ((lane >> 4) << 3);
}

// A fragment from LDS row-major [row][stride] (two contiguous 16B chunks / lane)
__device__ __forceinline__ v16h ldsA(const _Float16* base, int stride, int row0, int kbase, int lane) {
  const _Float16* p = base + (row0 + (lane & 15)) * stride + kbase + ((lane >> 4) << 3);
  v16h r;
#pragma unroll
  for (int i = 0; i < 8; i++) { r[i] = p[i]; r[i + 8] = p[i + 16]; }
  return r;
}

// B fragment from prepacked global weights (32B contiguous per lane)
__device__ __forceinline__ v16h ldB(const _Float16* pack, int frag, int lane) {
  const _Float16* p = pack + frag * 512 + lane * 16;
  v16h r;
#pragma unroll
  for (int i = 0; i < 16; i++) r[i] = p[i];
  return r;
}

// one 16x16 output tile, K=96 = 3 x 32 WMMA steps
__device__ __forceinline__ v8f gemm_tile(const _Float16* A, int astride, int row0,
                                         const _Float16* pack, int ntile, int lane) {
  v8f acc = {};
#pragma unroll
  for (int kk = 0; kk < 3; kk++) {
    v16h a = ldsA(A, astride, row0, kk * 32, lane);
    v16h b = ldB(pack, ntile * 3 + kk, lane);
    acc = __builtin_amdgcn_wmma_f32_16x16x32_f16(false, a, false, b, (short)0, acc, false, false);
  }
  return acc;
}

__device__ __forceinline__ bool row_valid(int m, int ty, int tx) {
  if (m >= SP) return false;
  int ly = m / SS, lx = m % SS;
  int gy = ty * TT - HALO + ly, gx = tx * TT - HALO + lx;
  return (gy >= 0) & (gy < FRW) & (gx >= 0) & (gx < FRW);
}

// ---- prepack: f32 weights -> f16 (WMMA-B-fragment order + conv-weight pack) ----
__global__ void elsa_prepack(const float* __restrict__ w_qk, const float* __restrict__ w_v,
                             const float* __restrict__ w_a2, const float* __restrict__ w_p,
                             const float* __restrict__ w_a1,
                             _Float16* __restrict__ wp) {
  int idx = blockIdx.x * 256 + threadIdx.x;
  if (idx >= WS_TOTAL) return;
  float val = 0.f;
  if (idx < WS_A2) {                       // fused qk|v : 288 cols x 96 k
    int f = idx >> 9, r = idx & 511;
    int ntile = f / 3, kk = f % 3, lane = r >> 4, e = r & 15;
    int col = ntile * 16 + (lane & 15);
    int k = kk * 32 + kfrag(lane, e);
    val = (col < 192) ? w_qk[col * CCH + k] : w_v[(col - 192) * CCH + k];
  } else if (idx < WS_P) {                 // a2, per-head padded to 64 cols
    int j = idx - WS_A2;
    int f = j >> 9, r = j & 511;
    int head = f / 12, rem = f % 12, ntile = rem / 3, kk = rem % 3;
    int lane = r >> 4, e = r & 15;
    int nloc = ntile * 16 + (lane & 15);
    int k = kk * 32 + kfrag(lane, e);
    if (nloc < K2) val = w_a2[(head * K2 + nloc) * CCH + k];
  } else if (idx < WS_A1) {                // out proj 96x96
    int j = idx - WS_P;
    int f = j >> 9, r = j & 511;
    int ntile = f / 3, kk = f % 3, lane = r >> 4, e = r & 15;
    int col = ntile * 16 + (lane & 15);
    int k = kk * 32 + kfrag(lane, e);
    val = w_p[col * CCH + k];
  } else {                                 // grouped conv: [oc][tap][ic] (ic fastest)
    int j = idx - WS_A1;
    int oc = j / (K2 * 4), r = j % (K2 * 4);
    int tap = r >> 2, ic = r & 3;
    val = w_a1[oc * 4 * K2 + ic * K2 + tap];
  }
  wp[idx] = (_Float16)val;
}

// ---- fused ELSA block: one 16x16 tile of one frame per workgroup ----
__global__ void __launch_bounds__(256, 1)
elsa_fused(const float* __restrict__ x, const float* __restrict__ ln_g, const float* __restrict__ ln_b,
           const float* __restrict__ b_qk, const float* __restrict__ b_v,
           const float* __restrict__ b_a1,
           const float* __restrict__ b_a2, const float* __restrict__ g_mul, const float* __restrict__ g_add,
           const float* __restrict__ b_p, const _Float16* __restrict__ wpack,
           float* __restrict__ out) {
  extern __shared__ _Float16 sm[];
  _Float16* xn      = sm + LDS_XN;
  _Float16* hb      = sm + LDS_H;
  _Float16* vb      = sm + LDS_V;
  _Float16* a_ls    = sm + LDS_A;
  _Float16* attn_ls = sm + LDS_ATTN;
  _Float16* agg     = sm + LDS_AGG;

  const int tid  = threadIdx.x;
  const int lane = tid & 31;
  const int wv   = tid >> 5;
  const int tile = blockIdx.x;  const int ty = tile / 6, tx = tile % 6;
  const int fr   = blockIdx.y;  const int bb = fr >> 3,  dd = fr & 7;
  const long frame_base = (long)bb * (96l * CH_STRIDE) + (long)dd * 9216;

  // warm L2 with packed weights while LN runs
  for (int off = tid * 64; off < WS_TOTAL; off += 256 * 64)
    __builtin_prefetch(wpack + off, 0, 1);

  // ---- Phase 1: load halo, LayerNorm over channels, store xn (f16) ----
  for (int p = tid; p < SROWS; p += 256) {
    _Float16* dst = xn + p * CCH;
    int ly = p / SS, lx = p % SS;
    int gy = ty * TT - HALO + ly, gx = tx * TT - HALO + lx;
    bool val = (p < SP) & (gy >= 0) & (gy < FRW) & (gx >= 0) & (gx < FRW);
    if (!val) {
      for (int c = 0; c < CCH; c++) dst[c] = (_Float16)0.f;
      continue;
    }
    const float* px = x + frame_base + (long)gy * FRW + gx;
    float s = 0.f, s2 = 0.f;
    for (int c = 0; c < CCH; c++) { float t = px[(long)c * CH_STRIDE]; s += t; s2 += t * t; }
    float mu = s * (1.f / CCH);
    float rs = rsqrtf(s2 * (1.f / CCH) - mu * mu + EPSL);
    for (int c = 0; c < CCH; c++) {
      float t = px[(long)c * CH_STRIDE];
      dst[c] = (_Float16)(((t - mu) * rs) * ln_g[c] + ln_b[c]);
    }
  }
  __syncthreads();

  // ---- Phase 2: fused qk|v projection GEMMs (WMMA). 31 m-tiles over 496 rows.
  //      jobs 0..185  : v      (col tiles 12..17)  -> vb
  //      jobs 186..371: (q,k) column-tile pairs    -> hb = (q+bq)(k+bk)*scale
  const _Float16* WQKV = wpack + WS_QKV;
  for (int job = wv; job < 372; job += 8) {
    if (job < 186) {
      int mt = job / 6, nt = job % 6;
      v8f acc = gemm_tile(xn, CCH, mt * 16, WQKV, 12 + nt, lane);
      int ch = nt * 16 + (lane & 15);
      float bv = b_v[ch];
#pragma unroll
      for (int r = 0; r < 8; r++) {
        int m = mt * 16 + r + ((lane >> 4) << 3);
        float vvv = acc[r] + bv;
        vb[m * CCH + ch] = (_Float16)(row_valid(m, ty, tx) ? vvv : 0.f);
      }
    } else {
      int t = job - 186;
      int mt = t / 6, j = t % 6;
      v8f q = gemm_tile(xn, CCH, mt * 16, WQKV, j, lane);
      v8f k = gemm_tile(xn, CCH, mt * 16, WQKV, 6 + j, lane);
      int ch = j * 16 + (lane & 15);
      float bq = b_qk[ch], bk = b_qk[96 + ch];
#pragma unroll
      for (int r = 0; r < 8; r++) {
        int m = mt * 16 + r + ((lane >> 4) << 3);
        float hv = (q[r] + bq) * (k[r] + bk) * SCL;
        hb[m * CCH + ch] = (_Float16)(row_valid(m, ty, tx) ? hv : 0.f);
      }
    }
  }
  __syncthreads();

  // ---- Phase 3: grouped 7x7 conv via packed f16 (v_pk_fma_f16), bias + exact GELU -> a
  {
    int p = tid, py = p >> 4, pxl = p & 15;
    const _Float16* wa1h = wpack + WS_A1;
    for (int oc = 0; oc < CCH; oc++) {
      int g4 = (oc >> 2) << 2;
      const h2* wr = (const h2*)(wa1h + oc * K2 * 4);   // uniform -> scalar loads
      h2 acc0 = {}, acc1 = {};
#pragma unroll 7
      for (int ij = 0; ij < K2; ij++) {
        int i = ij / KS, j = ij % KS;
        const h2* hp = (const h2*)(hb + ((py + i) * SS + (pxl + j)) * CCH + g4);
        acc0 = __builtin_elementwise_fma(wr[2 * ij],     hp[0], acc0);  // ic 0,1
        acc1 = __builtin_elementwise_fma(wr[2 * ij + 1], hp[1], acc1);  // ic 2,3
      }
      float accf = b_a1[oc] + (float)acc0[0] + (float)acc0[1]
                            + (float)acc1[0] + (float)acc1[1];
      float ge = 0.5f * accf * (1.f + erff(accf * 0.70710678118f));
      a_ls[p * CCH + oc] = (_Float16)ge;
    }
  }
  __syncthreads();

  // ---- Phase 4: per head: attn GEMM (WMMA, 49 taps padded to 64) + ghost,
  //               then 49-tap window aggregation (packed f16 FMA, 16 ch = 8 pairs).
  const _Float16* WA2 = wpack + WS_A2;
  for (int head = 0; head < NH; head++) {
    for (int job = wv; job < 64; job += 8) {
      int mt = job >> 2, nt = job & 3;
      v8f acc = gemm_tile(a_ls, CCH, mt * 16, WA2, head * 4 + nt, lane);
      int tap = nt * 16 + (lane & 15);
      bool tv = tap < K2;
      float ba = 0.f, gm = 0.f, ga = 0.f;
      if (tv) { int gi = head * K2 + tap; ba = b_a2[gi]; gm = g_mul[gi]; ga = g_add[gi]; }
#pragma unroll
      for (int r = 0; r < 8; r++) {
        int m = mt * 16 + r + ((lane >> 4) << 3);
        float av = tv ? ((acc[r] + ba) * gm + ga) : 0.f;
        attn_ls[m * 64 + tap] = (_Float16)av;
      }
    }
    __syncthreads();
    {
      int p = tid, py = p >> 4, pxl = p & 15;
      const _Float16* ap = attn_ls + p * 64;
      const int cb = head * HDIM;
      h2 acc[8];
#pragma unroll
      for (int cp = 0; cp < 8; cp++) acc[cp] = (h2){};
#pragma unroll 7
      for (int ij = 0; ij < K2; ij++) {
        int i = ij / KS, j = ij % KS;
        const h2* vrow = (const h2*)(vb + ((py + i) * SS + (pxl + j)) * CCH + cb);
        _Float16 a = ap[ij];
        h2 av; av[0] = a; av[1] = a;
#pragma unroll
        for (int cp = 0; cp < 8; cp++) acc[cp] = __builtin_elementwise_fma(av, vrow[cp], acc[cp]);
      }
      h2* arow = (h2*)(agg + p * CCH + cb);
#pragma unroll
      for (int cp = 0; cp < 8; cp++) arow[cp] = acc[cp];
    }
    __syncthreads();
  }

  // ---- Phase 5: output projection (WMMA) + bias + residual, store NCDHW ----
  const _Float16* WP = wpack + WS_P;
  for (int job = wv; job < 96; job += 8) {
    int mt = job / 6, nt = job % 6;
    v8f acc = gemm_tile(agg, CCH, mt * 16, WP, nt, lane);
    int ch = nt * 16 + (lane & 15);
    float bp = b_p[ch];
    const long cb = frame_base + (long)ch * CH_STRIDE;
#pragma unroll
    for (int r = 0; r < 8; r++) {
      int m = mt * 16 + r + ((lane >> 4) << 3);   // tile pixel index
      int gy = ty * TT + (m >> 4), gx = tx * TT + (m & 15);
      long idx = cb + (long)gy * FRW + gx;
      out[idx] = acc[r] + bp + x[idx];
    }
  }
}

extern "C" void kernel_launch(void* const* d_in, const int* in_sizes, int n_in,
                              void* d_out, int out_size, void* d_ws, size_t ws_size,
                              hipStream_t stream) {
  const float* x     = (const float*)d_in[0];
  const float* ln_g  = (const float*)d_in[1];
  const float* ln_b  = (const float*)d_in[2];
  const float* w_qk  = (const float*)d_in[3];
  const float* b_qk  = (const float*)d_in[4];
  const float* w_v   = (const float*)d_in[5];
  const float* b_v   = (const float*)d_in[6];
  const float* w_a1  = (const float*)d_in[7];
  const float* b_a1  = (const float*)d_in[8];
  const float* w_a2  = (const float*)d_in[9];
  const float* b_a2  = (const float*)d_in[10];
  const float* g_mul = (const float*)d_in[11];
  const float* g_add = (const float*)d_in[12];
  const float* w_p   = (const float*)d_in[13];
  const float* b_p   = (const float*)d_in[14];
  _Float16* wp = (_Float16*)d_ws;

  elsa_prepack<<<(WS_TOTAL + 255) / 256, 256, 0, stream>>>(w_qk, w_v, w_a2, w_p, w_a1, wp);

  dim3 grid(36, 16);  // 6x6 tiles, 16 frames
  elsa_fused<<<grid, 256, SMEM_BYTES, stream>>>(x, ln_g, ln_b, b_qk, b_v, b_a1,
                                                b_a2, g_mul, g_add, b_p, wp,
                                                (float*)d_out);
}
